// MEGNET_45810121179807
// MI455X (gfx1250) — matrix operand order repaired and checked by reading, not compiled, not run on hardware
//
#include <hip/hip_runtime.h>
#include <math.h>

// ---------------------------------------------------------------------------
// MEGNet forward for MI455X (gfx1250, wave32).
// All dense GEMMs on v_wmma_f32_16x16x32_f16 (f16 operands, f32 accumulate).
// - Weights pre-packed to f16 WMMA B-fragment layout => B fetch is one
//   unconditional 32B vector load per (chunk, ntile).
// - Stage-1 A fragments built straight from global with float2 loads;
//   concat segment selection is compile-time per 32-wide K chunk; row
//   bounds handled by index clamping (epilogue masks stores) => no EXEC
//   divergence ahead of the WMMAs.
// - Stages 2/3 relayout activations through per-wave LDS (f16).
// - All transcendentals use branchless hardware v_exp/v_log/v_rcp forms.
// ---------------------------------------------------------------------------

typedef __attribute__((ext_vector_type(16))) _Float16 v16h;
typedef __attribute__((ext_vector_type(2)))  _Float16 h2v;
typedef __attribute__((ext_vector_type(8)))  float    v8f;

// Branchless softplus: max(x,0) + log(1+exp(-|x|)); log arg in [1,2].
__device__ inline float sp_f(float x) {
  float t = __expf(-fabsf(x));
  return fmaxf(x, 0.f) + __logf(1.f + t);
}
__device__ inline float sig_f(float x) {
  return __builtin_amdgcn_rcpf(1.f + __expf(-x));
}
__device__ inline float tanh_f(float x) {
  float e = __expf(2.f * x);
  return 1.f - 2.f * __builtin_amdgcn_rcpf(e + 1.f);
}

template <int NT>
__device__ inline void zero_acc(v8f* acc) {
#pragma unroll
  for (int nt = 0; nt < NT; ++nt)
#pragma unroll
    for (int r = 0; r < 8; ++r) acc[nt][r] = 0.f;
}

// One K-chunk of WMMAs against pre-packed B fragments.
// Packed layout: ((c*NT + nt)*32 + lane)*16 + i  (i = half index 0..15).
template <int NT>
__device__ inline void wmma_nt(const v16h& a, const _Float16* Wp, int c, v8f* acc, int lane) {
#pragma unroll
  for (int nt = 0; nt < NT; ++nt) {
    v16h b = *(const v16h*)(Wp + ((c * NT + nt) * 32 + lane) * 16);
    acc[nt] = __builtin_amdgcn_wmma_f32_16x16x32_f16(
        false, a, false, b, (short)0, acc[nt], false, false);
  }
}

// A fragment (16x32 f16, ISA 7.12.2 layout) from a global f32 row pointer.
// rp must already include +kh8 (lane-half K offset). 8x float2 loads.
__device__ inline v16h a_rowptr(const float* rp, float scale) {
  v16h a;
#pragma unroll
  for (int r = 0; r < 8; ++r) {
    int kb = ((r & 3) << 1) + ((r >> 2) << 4);
    float2 w = *(const float2*)(rp + kb);
    a[2 * r]     = (_Float16)(w.x * scale);
    a[2 * r + 1] = (_Float16)(w.y * scale);
  }
  return a;
}

// K=16 variant (ffv0 input): regs r4..r7 (k>=16) are compile-time zero.
__device__ inline v16h a_rowptr16(const float* rp) {
  v16h a;
#pragma unroll
  for (int r = 0; r < 4; ++r) {
    int kb = (r & 3) << 1;
    float2 w = *(const float2*)(rp + kb);
    a[2 * r]     = (_Float16)w.x;
    a[2 * r + 1] = (_Float16)w.y;
  }
#pragma unroll
  for (int r = 4; r < 8; ++r) { a[2 * r] = (_Float16)0.f; a[2 * r + 1] = (_Float16)0.f; }
  return a;
}

// RBF expansion chunk: pure ALU, k>=100 masked with cndmask (no memory).
__device__ inline v16h a_rbf(float d, int kc, int kh8) {
  v16h a;
#pragma unroll
  for (int r = 0; r < 8; ++r) {
    int kb = ((r & 3) << 1) + ((r >> 2) << 4);
#pragma unroll
    for (int h = 0; h < 2; ++h) {
      int k = kc + kh8 + kb + h;
      float c = 5.f * (float)k / 99.f;
      float dd = d - c;
      float v = __expf(-4.f * dd * dd);
      a[2 * r + h] = (_Float16)((k < 100) ? v : 0.f);
    }
  }
  return a;
}

// A fragment from LDS f16 tile (row stride 64 halfs); packed-pair ds_load_b32.
__device__ inline v16h a_from_lds(const _Float16* rp) {
  v16h a;
#pragma unroll
  for (int r = 0; r < 8; ++r) {
    int kb = ((r & 3) << 1) + ((r >> 2) << 4);
    h2v p = *(const h2v*)(rp + kb);
    a[2 * r]     = p[0];
    a[2 * r + 1] = p[1];
  }
  return a;
}

// C/D layout: lane holds col n0=lane&15, rows r+8*(lane>>4).
template <int NT>
__device__ inline void act_to_lds(_Float16* dst, const v8f* acc, const float* bias, int lane) {
  int n0 = lane & 15, mh = (lane >> 4) * 8;
#pragma unroll
  for (int r = 0; r < 8; ++r)
#pragma unroll
    for (int nt = 0; nt < NT; ++nt) {
      float v = sp_f(acc[nt][r] + bias[nt * 16 + n0]);
      dst[(mh + r) * 64 + nt * 16 + n0] = (_Float16)v;
    }
}

template <int NT, int CH>
__device__ inline void stage_lds(const _Float16* buf, const _Float16* Wp, v8f* acc, int lane) {
  int m = lane & 15, kh8 = (lane >> 4) << 3;
  zero_acc<NT>(acc);
#pragma unroll
  for (int c = 0; c < CH; ++c) {
    v16h a = a_from_lds(buf + m * 64 + c * 32 + kh8);
    wmma_nt<NT>(a, Wp, c, acc, lane);
  }
}

template <int NT, bool ACCUM, bool SCAT, bool FSP>
__device__ inline void emit_out(float* out, float* accum, const int* sidx, long rows,
                                long row0, const v8f* acc, const float* bias, int lane) {
  int n0 = lane & 15, mh = (lane >> 4) * 8;
#pragma unroll
  for (int r = 0; r < 8; ++r) {
    long row = row0 + mh + r;
    if (row < rows) {
      int sg = SCAT ? sidx[row] : 0;
#pragma unroll
      for (int nt = 0; nt < NT; ++nt) {
        float v = acc[nt][r] + bias[nt * 16 + n0];
        if (FSP) v = sp_f(v);
        long o = row * (NT * 16) + nt * 16 + n0;
        if (ACCUM) out[o] += v; else out[o] = v;
        if (SCAT) atomicAdd(&accum[(long)sg * (NT * 16) + nt * 16 + n0], v);
      }
    }
  }
}

enum { LD_DENSE32 = 0, LD_DENSE16 = 1, LD_RBF = 2, LD_ZERO = 3,
       LD_EDGECAT = 4, LD_NODECAT = 5, LD_UCAT = 6 };

struct FArgs {
  const float *in0, *in1, *in2;
  const int *gi0, *gi1, *gi2;                 // src, dst, batch
  const _Float16 *Wp1, *Wp2, *Wp3;            // pre-packed B fragments
  const float *B1, *B2, *B3;                  // biases (f32)
  float* out; float* accum; const int* sidx;
  const float* dvec; const float* cnt;
  long rows;
};

template <int MODE, int NST, int NT3, bool ACCUM, bool SCAT>
__global__ __launch_bounds__(256) void k_fused(FArgs A) {
  __shared__ _Float16 smem[8 * 2048];          // per wave: 2x (16x64) f16
  int lane = threadIdx.x & 31;
  int wave = threadIdx.x >> 5;
  long row0 = ((long)blockIdx.x * 8 + wave) * 16;
  if (row0 >= A.rows) return;                  // wave-uniform
  _Float16* bufA = smem + wave * 2048;
  _Float16* bufB = bufA + 1024;
  int m = lane & 15;
  int kh8 = (lane >> 4) << 3;
  long row = row0 + m;
  long rowc = row < A.rows ? row : (A.rows - 1);  // clamp: no divergence pre-WMMA

  constexpr int NT1 = (NST == 1) ? NT3 : 4;
  v8f acc1[NT1];
  zero_acc<NT1>(acc1);

  if constexpr (MODE == LD_DENSE32) {
    v16h a = a_rowptr(A.in0 + rowc * 32 + kh8, 1.f);
    wmma_nt<NT1>(a, A.Wp1, 0, acc1, lane);
  } else if constexpr (MODE == LD_DENSE16) {
    v16h a = a_rowptr16(A.in0 + rowc * 16 + kh8);
    wmma_nt<NT1>(a, A.Wp1, 0, acc1, lane);
  } else if constexpr (MODE == LD_RBF) {
    float d = A.dvec[rowc];
#pragma unroll
    for (int c = 0; c < 4; ++c) {
      v16h a = a_rbf(d, c * 32, kh8);
      wmma_nt<NT1>(a, A.Wp1, c, acc1, lane);
    }
  } else if constexpr (MODE == LD_EDGECAT) {   // concat(v[src], v[dst], u[batch[src]], e)
    long s = A.gi0[rowc], t = A.gi1[rowc];
    long bb = A.gi2[s];
    { v16h a = a_rowptr(A.in0 + s * 32 + kh8, 1.f);    wmma_nt<NT1>(a, A.Wp1, 0, acc1, lane); }
    { v16h a = a_rowptr(A.in0 + t * 32 + kh8, 1.f);    wmma_nt<NT1>(a, A.Wp1, 1, acc1, lane); }
    { v16h a = a_rowptr(A.in1 + bb * 32 + kh8, 1.f);   wmma_nt<NT1>(a, A.Wp1, 2, acc1, lane); }
    { v16h a = a_rowptr(A.in2 + rowc * 32 + kh8, 1.f); wmma_nt<NT1>(a, A.Wp1, 3, acc1, lane); }
  } else if constexpr (MODE == LD_NODECAT) {   // concat(v, etv, u[batch])
    long bb = A.gi2[rowc];
    { v16h a = a_rowptr(A.in0 + rowc * 32 + kh8, 1.f); wmma_nt<NT1>(a, A.Wp1, 0, acc1, lane); }
    { v16h a = a_rowptr(A.in1 + rowc * 32 + kh8, 1.f); wmma_nt<NT1>(a, A.Wp1, 1, acc1, lane); }
    { v16h a = a_rowptr(A.in2 + bb * 32 + kh8, 1.f);   wmma_nt<NT1>(a, A.Wp1, 2, acc1, lane); }
  } else if constexpr (MODE == LD_UCAT) {      // concat(ue_sum/cnt, uv_sum/cnt, u)
    float inv = __builtin_amdgcn_rcpf(fmaxf(A.cnt[rowc], 1.f));
    { v16h a = a_rowptr(A.in0 + rowc * 32 + kh8, inv); wmma_nt<NT1>(a, A.Wp1, 0, acc1, lane); }
    { v16h a = a_rowptr(A.in1 + rowc * 32 + kh8, inv); wmma_nt<NT1>(a, A.Wp1, 1, acc1, lane); }
    { v16h a = a_rowptr(A.in2 + rowc * 32 + kh8, 1.f); wmma_nt<NT1>(a, A.Wp1, 2, acc1, lane); }
  }
  // LD_ZERO: A == 0 -> acc1 stays 0 (0 @ W == 0), bias/softplus applied below.

  if constexpr (NST == 1) {
    emit_out<NT3, ACCUM, SCAT, false>(A.out, A.accum, A.sidx, A.rows, row0, acc1, A.B1, lane);
  } else if constexpr (NST == 2) {
    act_to_lds<4>(bufA, acc1, A.B1, lane);
    v8f acc2[NT3];
    stage_lds<NT3, 2>(bufA, A.Wp2, acc2, lane);
    emit_out<NT3, ACCUM, SCAT, true>(A.out, A.accum, A.sidx, A.rows, row0, acc2, A.B2, lane);
  } else {
    act_to_lds<4>(bufA, acc1, A.B1, lane);
    v8f acc2[4];
    stage_lds<4, 2>(bufA, A.Wp2, acc2, lane);
    act_to_lds<4>(bufB, acc2, A.B2, lane);
    v8f acc3[NT3];
    stage_lds<NT3, 2>(bufB, A.Wp3, acc3, lane);
    emit_out<NT3, ACCUM, SCAT, true>(A.out, A.accum, A.sidx, A.rows, row0, acc3, A.B3, lane);
  }
}

// ---- weight pre-pack: W[K,NOUT] f32 -> f16 WMMA B fragments, K zero-padded --
__global__ void k_packB(const float* W, _Float16* out, int K, int NOUT, int CH) {
  int total = CH * (NOUT >> 4) * 512;
  int idx = blockIdx.x * blockDim.x + threadIdx.x;
  if (idx >= total) return;
  int i    = idx & 15;
  int lane = (idx >> 4) & 31;
  int nt   = (idx >> 9) % (NOUT >> 4);
  int c    = idx / (512 * (NOUT >> 4));
  int k    = c * 32 + ((i >> 3) << 4) + ((lane >> 4) << 3) + (i & 7);
  int col  = nt * 16 + (lane & 15);
  out[idx] = (_Float16)((k < K) ? W[(long)k * NOUT + col] : 0.f);
}

// ------------------------------ small kernels ------------------------------

__global__ void k_prep_edge(const int* ei, const int* batch, const float* eattr,
                            float* dvec, int* ebatch, float* deg, long E) {
  long i = (long)blockIdx.x * blockDim.x + threadIdx.x;
  if (i >= E) return;
  float ax = eattr[i * 3], ay = eattr[i * 3 + 1], az = eattr[i * 3 + 2];
  dvec[i] = sqrtf(ax * ax + ay * ay + az * az);
  int d = ei[E + i];
  ebatch[i] = batch[d];
  atomicAdd(&deg[d], 1.f);
}

__global__ void k_cnt(const int* batch, float* cnt, long N) {
  long i = (long)blockIdx.x * blockDim.x + threadIdx.x;
  if (i >= N) return;
  atomicAdd(&cnt[batch[i]], 1.f);
}

__global__ void k_emb(const float* x, const float* W, const float* b, float* v16, long N) {
  long i = (long)blockIdx.x * blockDim.x + threadIdx.x;
  if (i >= N * 16) return;
  long n = i >> 4; int c = (int)(i & 15);
  float s = b[c];
#pragma clang loop unroll_count(4)
  for (int k = 0; k < 92; ++k) s += x[n * 92 + k] * W[k * 16 + c];
  v16[i] = s;
}

__global__ void k_etvnorm(const float* etv_sum, const float* deg, const int* batch,
                          float* etv, float* ue_sum, long N) {
  __shared__ float lu[64 * 32];
  for (int t = threadIdx.x; t < 64 * 32; t += blockDim.x) lu[t] = 0.f;
  __syncthreads();
  long i = (long)blockIdx.x * blockDim.x + threadIdx.x;
  if (i < N * 32) {
    long n = i >> 5; int c = (int)(i & 31);
    float v = etv_sum[i] / fmaxf(deg[n], 1.f);
    etv[i] = v;
    atomicAdd(&lu[batch[n] * 32 + c], v);
  }
  __syncthreads();
  for (int t = threadIdx.x; t < 64 * 32; t += blockDim.x)
    if (lu[t] != 0.f) atomicAdd(&ue_sum[t], lu[t]);
}

// ------------------------------ set2set ------------------------------------

__global__ void k_lstm(const float* Wih, const float* Whh, const float* bih, const float* bhh,
                       const float* qs, float* hh, float* cc,
                       float* m, float* sseg, float* rbuf) {
  int b = threadIdx.x;
  if (b >= 64) return;
  float g[64];
#pragma clang loop unroll(disable)
  for (int j = 0; j < 64; ++j) {
    float s = bih[j] + bhh[j];
#pragma clang loop unroll_count(4)
    for (int k = 0; k < 32; ++k) s += qs[b * 32 + k] * Wih[j * 32 + k];
#pragma clang loop unroll_count(4)
    for (int k = 0; k < 16; ++k) s += hh[b * 16 + k] * Whh[j * 16 + k];
    g[j] = s;
  }
#pragma clang loop unroll(disable)
  for (int k = 0; k < 16; ++k) {
    float ig = sig_f(g[k]);
    float fg = sig_f(g[16 + k]);
    float gg = tanh_f(g[32 + k]);
    float og = sig_f(g[48 + k]);
    float c  = fg * cc[b * 16 + k] + ig * gg;
    cc[b * 16 + k] = c;
    hh[b * 16 + k] = og * tanh_f(c);
  }
  m[b] = -INFINITY;
  sseg[b] = 0.f;
  for (int k = 0; k < 16; ++k) rbuf[b * 16 + k] = 0.f;
}

__device__ inline void atomicMaxFBits(float* addr, float v) {
  if (v >= 0.f) atomicMax((int*)addr, __float_as_int(v));
  else          atomicMin((unsigned int*)addr, __float_as_uint(v));
}

__global__ void k_score(const float* h, const int* idx, const float* hh,
                        float* score, float* m, long rows) {
  __shared__ float lm[64];
  if (threadIdx.x < 64) lm[threadIdx.x] = -INFINITY;
  __syncthreads();
  long i = (long)blockIdx.x * blockDim.x + threadIdx.x;
  if (i < rows) {
    int b = idx[i];
    float s = 0.f;
#pragma unroll
    for (int c = 0; c < 16; ++c) s += h[i * 16 + c] * hh[b * 16 + c];
    score[i] = s;
    atomicMaxFBits(&lm[b], s);
  }
  __syncthreads();
  if (threadIdx.x < 64 && isfinite(lm[threadIdx.x]))
    atomicMaxFBits(&m[threadIdx.x], lm[threadIdx.x]);
}

__global__ void k_expsum(const int* idx, const float* m, const float* score,
                         float* abuf, float* sseg, long rows) {
  __shared__ float ls[64];
  if (threadIdx.x < 64) ls[threadIdx.x] = 0.f;
  __syncthreads();
  long i = (long)blockIdx.x * blockDim.x + threadIdx.x;
  if (i < rows) {
    int b = idx[i];
    float mm = m[b];
    if (!isfinite(mm)) mm = 0.f;
    float a = __expf(score[i] - mm);
    abuf[i] = a;
    atomicAdd(&ls[b], a);
  }
  __syncthreads();
  if (threadIdx.x < 64 && ls[threadIdx.x] != 0.f) atomicAdd(&sseg[threadIdx.x], ls[threadIdx.x]);
}

__global__ void k_attn(const float* h, const int* idx, const float* abuf,
                       const float* sseg, float* rbuf, long rows) {
  __shared__ float lr[64 * 16];
  for (int t = threadIdx.x; t < 64 * 16; t += blockDim.x) lr[t] = 0.f;
  __syncthreads();
  long i = (long)blockIdx.x * blockDim.x + threadIdx.x;
  if (i < rows) {
    int b = idx[i];
    float w = abuf[i] * __builtin_amdgcn_rcpf(fmaxf(sseg[b], 1e-16f));
#pragma unroll
    for (int c = 0; c < 16; ++c) atomicAdd(&lr[b * 16 + c], w * h[i * 16 + c]);
  }
  __syncthreads();
  for (int t = threadIdx.x; t < 64 * 16; t += blockDim.x)
    if (lr[t] != 0.f) atomicAdd(&rbuf[t], lr[t]);
}

__global__ void k_qstar(const float* hh, const float* rbuf, float* qs) {
  int b = threadIdx.x;
  if (b >= 64) return;
  for (int c = 0; c < 16; ++c) {
    qs[b * 32 + c]      = hh[b * 16 + c];
    qs[b * 32 + 16 + c] = rbuf[b * 16 + c];
  }
}

// ------------------------------ readout ------------------------------------

__global__ void k_hcat(const float* qn, const float* qe, const float* u, float* hcat) {
  int i = blockIdx.x * blockDim.x + threadIdx.x;
  if (i >= 64 * 96) return;
  int b = i / 96, c = i % 96;
  float v = (c < 32) ? qn[b * 32 + c] : (c < 64) ? qe[b * 32 + (c - 32)] : u[b * 32 + (c - 64)];
  hcat[i] = v;
}

__global__ void k_fcout(const float* hcat, const float* W0, const float* b0,
                        const float* W1, const float* b1, const float* W2, const float* b2,
                        float* out) {
  int b = threadIdx.x;
  if (b >= 64) return;
  float h1[32], h2[16];
#pragma clang loop unroll(disable)
  for (int j = 0; j < 32; ++j) {
    float s = b0[j];
#pragma clang loop unroll_count(4)
    for (int k = 0; k < 96; ++k) s += hcat[b * 96 + k] * W0[k * 32 + j];
    h1[j] = sp_f(s);
  }
#pragma clang loop unroll(disable)
  for (int j = 0; j < 16; ++j) {
    float s = b1[j];
#pragma clang loop unroll_count(4)
    for (int k = 0; k < 32; ++k) s += h1[k] * W1[k * 16 + j];
    h2[j] = sp_f(s);
  }
#pragma clang loop unroll(disable)
  for (int j = 0; j < 9; ++j) {
    float s = b2[j];
#pragma clang loop unroll_count(4)
    for (int k = 0; k < 16; ++k) s += h2[k] * W2[k * 9 + j];
    out[b * 9 + j] = s;
  }
}

// ------------------------------ host side ----------------------------------

static inline unsigned gdiv(long n, long d) { return (unsigned)((n + d - 1) / d); }

extern "C" void kernel_launch(void* const* d_in, const int* in_sizes, int n_in,
                              void* d_out, int out_size, void* d_ws, size_t ws_size,
                              hipStream_t stream) {
  (void)n_in; (void)out_size; (void)ws_size;
  // jax tree-leaf order (dict keys sorted): batch, edge_attr, edge_index, params..., x
  const long N = in_sizes[0];
  const long E = in_sizes[2] / 2;
  const int  B = 64;
  const int*   batch = (const int*)d_in[0];
  const float* eattr = (const float*)d_in[1];
  const int*   ei    = (const int*)d_in[2];
  const int*   src   = ei;
  const int*   dst   = ei + E;
  auto F = [&](int i) { return (const float*)d_in[i]; };
  const float* x = F(113);
  // params: 3-4 edge_linear; 5-8 edge_s2s; 9-10 emb; 11-16 fc_out; 17-20 ffe0;
  // 21-24 ffe1; 25-28 ffe2; 29-32 ffu0; 33-36 ffu1; 37-40 ffu2; 41-44 ffv0;
  // 45-48 ffv1; 49-52 ffv2; 53-70 meg1(e,u,v); 71-88 meg2; 89-106 meg3;
  // 107-108 node_linear; 109-112 node_s2s.

  // workspace carve
  char* basep = (char*)d_ws;
  size_t off = 0;
  auto alloc = [&](size_t bytes) -> void* {
    void* p = basep + off;
    off = (off + bytes + 255) & ~(size_t)255;
    return p;
  };
  float* E_state = (float*)alloc((size_t)E * 32 * 4);
  float* E_ff    = (float*)alloc((size_t)E * 32 * 4);
  float* E16     = (float*)alloc((size_t)E * 16 * 4);
  float* dvec    = (float*)alloc((size_t)E * 4);
  float* score   = (float*)alloc((size_t)E * 4);
  float* abuf    = (float*)alloc((size_t)E * 4);
  int*   ebatch  = (int*)alloc((size_t)E * 4);
  float* V16     = (float*)alloc((size_t)N * 16 * 4);
  float* V_state = (float*)alloc((size_t)N * 32 * 4);
  float* V_ff    = (float*)alloc((size_t)N * 32 * 4);
  float* etv_sum = (float*)alloc((size_t)N * 32 * 4);
  float* etv     = (float*)alloc((size_t)N * 32 * 4);
  float* VN16    = (float*)alloc((size_t)N * 16 * 4);
  float* deg     = (float*)alloc((size_t)N * 4);
  float* cnt     = (float*)alloc(B * 4);
  float* U_state = (float*)alloc(B * 32 * 4);
  float* U_ff    = (float*)alloc(B * 32 * 4);
  float* ue_sum  = (float*)alloc(B * 32 * 4);
  float* uv_sum  = (float*)alloc(B * 32 * 4);
  float* hh      = (float*)alloc(B * 16 * 4);
  float* ccb     = (float*)alloc(B * 16 * 4);
  float* qsN     = (float*)alloc(B * 32 * 4);
  float* qsE     = (float*)alloc(B * 32 * 4);
  float* mseg    = (float*)alloc(B * 4);
  float* sseg    = (float*)alloc(B * 4);
  float* rbuf    = (float*)alloc(B * 16 * 4);
  float* hcat    = (float*)alloc(B * 96 * 4);
  _Float16* wpool = (_Float16*)alloc(200000 * 2);   // packed weights (~320KB used)

  // ---- pre-pack all GEMM weights into WMMA B-fragment layout ----
  size_t woff = 0;
  auto pack = [&](const float* W, int K, int NOUT) -> const _Float16* {
    int CH = (K + 31) / 32;
    _Float16* dstp = wpool + woff;
    int total = CH * (NOUT / 16) * 512;
    woff += (size_t)total;
    k_packB<<<gdiv(total, 256), 256, 0, stream>>>(W, dstp, K, NOUT, CH);
    return dstp;
  };
  const _Float16 *Pffv[3][2], *Pffe[3][2], *Pffu[3][2];
  for (int m = 0; m < 3; ++m) {
    Pffv[m][0] = pack(F(41 + 4 * m), m == 0 ? 16 : 32, 64);
    Pffv[m][1] = pack(F(43 + 4 * m), 64, 32);
    Pffe[m][0] = pack(F(17 + 4 * m), m == 0 ? 100 : 32, 64);
    Pffe[m][1] = pack(F(19 + 4 * m), 64, 32);
    Pffu[m][0] = (m == 0) ? (const _Float16*)nullptr : pack(F(29 + 4 * m), 32, 64);
    Pffu[m][1] = pack(F(31 + 4 * m), 64, 32);
  }
  const _Float16* Pmeg[3][3][3];  // [meg][e,u,v][stage]
  for (int m = 0; m < 3; ++m) {
    int pb = 53 + m * 18;
    Pmeg[m][0][0] = pack(F(pb), 128, 64);       // e
    Pmeg[m][0][1] = pack(F(pb + 2), 64, 64);
    Pmeg[m][0][2] = pack(F(pb + 4), 64, 32);
    Pmeg[m][1][0] = pack(F(pb + 6), 96, 64);    // u
    Pmeg[m][1][1] = pack(F(pb + 8), 64, 64);
    Pmeg[m][1][2] = pack(F(pb + 10), 64, 32);
    Pmeg[m][2][0] = pack(F(pb + 12), 96, 64);   // v
    Pmeg[m][2][1] = pack(F(pb + 14), 64, 64);
    Pmeg[m][2][2] = pack(F(pb + 16), 64, 32);
  }
  const _Float16* Pnlin = pack(F(107), 32, 16);
  const _Float16* Pelin = pack(F(3), 32, 16);

  auto gw = [&](long rows) { return gdiv(rows, 128); };  // 8 waves x 16 rows / block

  // ---- prep ----
  hipMemsetAsync(deg, 0, (size_t)N * 4, stream);
  hipMemsetAsync(cnt, 0, B * 4, stream);
  k_prep_edge<<<gdiv(E, 256), 256, 0, stream>>>(ei, batch, eattr, dvec, ebatch, deg, E);
  k_cnt<<<gdiv(N, 256), 256, 0, stream>>>(batch, cnt, N);
  k_emb<<<gdiv(N * 16, 256), 256, 0, stream>>>(x, F(9), F(10), V16, N);

  // ---- initial FF blocks ----
  { FArgs A{}; A.in0 = V16; A.Wp1 = Pffv[0][0]; A.B1 = F(42); A.Wp2 = Pffv[0][1]; A.B2 = F(44);
    A.out = V_state; A.rows = N;
    k_fused<LD_DENSE16, 2, 2, false, false><<<gw(N), 256, 0, stream>>>(A); }
  { FArgs A{}; A.dvec = dvec; A.Wp1 = Pffe[0][0]; A.B1 = F(18); A.Wp2 = Pffe[0][1]; A.B2 = F(20);
    A.out = E_state; A.rows = E;
    k_fused<LD_RBF, 2, 2, false, false><<<gw(E), 256, 0, stream>>>(A); }
  { FArgs A{}; A.B1 = F(30); A.Wp2 = Pffu[0][1]; A.B2 = F(32);
    A.out = U_state; A.rows = 64;
    k_fused<LD_ZERO, 2, 2, false, false><<<gw(64), 256, 0, stream>>>(A); }

  // ---- 3 MEGNet blocks ----
  for (int mb = 0; mb < 3; ++mb) {
    const float *vin, *ein, *uin;
    if (mb == 0) { vin = V_state; ein = E_state; uin = U_state; }
    else {
      { FArgs A{}; A.in0 = V_state; A.Wp1 = Pffv[mb][0]; A.B1 = F(42 + 4 * mb);
        A.Wp2 = Pffv[mb][1]; A.B2 = F(44 + 4 * mb); A.out = V_ff; A.rows = N;
        k_fused<LD_DENSE32, 2, 2, false, false><<<gw(N), 256, 0, stream>>>(A); }
      { FArgs A{}; A.in0 = E_state; A.Wp1 = Pffe[mb][0]; A.B1 = F(18 + 4 * mb);
        A.Wp2 = Pffe[mb][1]; A.B2 = F(20 + 4 * mb); A.out = E_ff; A.rows = E;
        k_fused<LD_DENSE32, 2, 2, false, false><<<gw(E), 256, 0, stream>>>(A); }
      { FArgs A{}; A.in0 = U_state; A.Wp1 = Pffu[mb][0]; A.B1 = F(30 + 4 * mb);
        A.Wp2 = Pffu[mb][1]; A.B2 = F(32 + 4 * mb); A.out = U_ff; A.rows = 64;
        k_fused<LD_DENSE32, 2, 2, false, false><<<gw(64), 256, 0, stream>>>(A); }
      vin = V_ff; ein = E_ff; uin = U_ff;
    }
    const int pb = 53 + mb * 18;
    hipMemsetAsync(etv_sum, 0, (size_t)N * 32 * 4, stream);
    hipMemsetAsync(ue_sum, 0, B * 32 * 4, stream);
    hipMemsetAsync(uv_sum, 0, B * 32 * 4, stream);
    // edge MLP: E_state += e_p; etv_sum[dst] += e_p
    { FArgs A{}; A.in0 = vin; A.in1 = uin; A.in2 = ein;
      A.gi0 = src; A.gi1 = dst; A.gi2 = batch;
      A.Wp1 = Pmeg[mb][0][0]; A.B1 = F(pb + 1);
      A.Wp2 = Pmeg[mb][0][1]; A.B2 = F(pb + 3);
      A.Wp3 = Pmeg[mb][0][2]; A.B3 = F(pb + 5);
      A.out = E_state; A.accum = etv_sum; A.sidx = dst; A.rows = E;
      k_fused<LD_EDGECAT, 3, 2, true, true><<<gw(E), 256, 0, stream>>>(A); }
    k_etvnorm<<<gdiv(N * 32, 256), 256, 0, stream>>>(etv_sum, deg, batch, etv, ue_sum, N);
    // node MLP: V_state += v_p; uv_sum[batch] += v_p
    { FArgs A{}; A.in0 = vin; A.in1 = etv; A.in2 = uin; A.gi2 = batch;
      A.Wp1 = Pmeg[mb][2][0]; A.B1 = F(pb + 13);
      A.Wp2 = Pmeg[mb][2][1]; A.B2 = F(pb + 15);
      A.Wp3 = Pmeg[mb][2][2]; A.B3 = F(pb + 17);
      A.out = V_state; A.accum = uv_sum; A.sidx = batch; A.rows = N;
      k_fused<LD_NODECAT, 3, 2, true, true><<<gw(N), 256, 0, stream>>>(A); }
    // u MLP: U_state += u_p
    { FArgs A{}; A.in0 = ue_sum; A.in1 = uv_sum; A.in2 = uin; A.cnt = cnt;
      A.Wp1 = Pmeg[mb][1][0]; A.B1 = F(pb + 7);
      A.Wp2 = Pmeg[mb][1][1]; A.B2 = F(pb + 9);
      A.Wp3 = Pmeg[mb][1][2]; A.B3 = F(pb + 11);
      A.out = U_state; A.rows = 64;
      k_fused<LD_UCAT, 3, 2, true, false><<<gw(64), 256, 0, stream>>>(A); }
  }

  // ---- node/edge linear (32 -> 16, no activation) ----
  { FArgs A{}; A.in0 = V_state; A.Wp1 = Pnlin; A.B1 = F(108);
    A.out = VN16; A.rows = N;
    k_fused<LD_DENSE32, 1, 1, false, false><<<gw(N), 256, 0, stream>>>(A); }
  { FArgs A{}; A.in0 = E_state; A.Wp1 = Pelin; A.B1 = F(4);
    A.out = E16; A.rows = E;
    k_fused<LD_DENSE32, 1, 1, false, false><<<gw(E), 256, 0, stream>>>(A); }

  // ---- set2set (nodes then edges) ----
  auto set2set = [&](const float* h, const int* idx, long rows,
                     const float* Wih, const float* Whh, const float* bih, const float* bhh,
                     float* qs) {
    hipMemsetAsync(hh, 0, B * 16 * 4, stream);
    hipMemsetAsync(ccb, 0, B * 16 * 4, stream);
    hipMemsetAsync(qs, 0, B * 32 * 4, stream);
    for (int s = 0; s < 3; ++s) {
      k_lstm<<<1, 64, 0, stream>>>(Wih, Whh, bih, bhh, qs, hh, ccb, mseg, sseg, rbuf);
      k_score<<<gdiv(rows, 256), 256, 0, stream>>>(h, idx, hh, score, mseg, rows);
      k_expsum<<<gdiv(rows, 256), 256, 0, stream>>>(idx, mseg, score, abuf, sseg, rows);
      k_attn<<<gdiv(rows, 256), 256, 0, stream>>>(h, idx, abuf, sseg, rbuf, rows);
      k_qstar<<<1, 64, 0, stream>>>(hh, rbuf, qs);
    }
  };
  set2set(VN16, batch, N, F(109), F(110), F(111), F(112), qsN);
  set2set(E16, ebatch, E, F(5), F(6), F(7), F(8), qsE);

  // ---- readout ----
  k_hcat<<<gdiv(64 * 96, 256), 256, 0, stream>>>(qsN, qsE, U_state, hcat);
  k_fcout<<<1, 64, 0, stream>>>(hcat, F(11), F(12), F(13), F(14), F(15), F(16),
                                (float*)d_out);
}
